// RegionAttrEncoder_22196390985919
// MI455X (gfx1250) — matrix operand order repaired and checked by reading, not compiled, not run on hardware
//
#include <hip/hip_runtime.h>
#include <math.h>

// ---------------------------------------------------------------------------
// CDNA5 / gfx1250 RegionAttrEncoder forward.
// Convs = implicit GEMM on v_wmma_f32_16x16x32_f16 (wave32).
// Activations keep f16 NHWC mirrors (channel stride padded to 32) so each
// lane's 16-half B fragment is ONE aligned 32B vector load. Weights are
// pre-packed into A-fragment order with K = (kh*ks+kw)*Cinp + ci.
// ---------------------------------------------------------------------------

typedef __attribute__((ext_vector_type(16))) _Float16 v16h;
typedef __attribute__((ext_vector_type(8)))  float    v8f;

// ---------------------------------------------------------------------------
// Weight packing into A-fragment order (16x32 f16 layout):
//   lane&15 = M row, K: e -> akb+e, akb=16*(lane>=16)
//   K index = s*Cinp + ci  (s = kh*ks+kw, Cinp = Cin rounded up to 32)
//   mode 0/1: OIHW -> (co*Cin + ci)*ks2 + s
//   mode 2  : torch convT [Cin,Cout,3,3] flipped -> ci*Cout*9 + co*9 + (8-s)
// grid = (nChunk, Cout/16), block = 32
// ---------------------------------------------------------------------------
__global__ void pack_w_kernel(const float* __restrict__ W, _Float16* __restrict__ P,
                              int Cout, int Cin, int ks, int mode) {
  const int ks2    = ks * ks;
  const int Cinp   = (Cin + 31) & ~31;
  const int nCper  = Cinp >> 5;
  const int nChunk = ks2 * nCper;
  const int chunk  = blockIdx.x;
  const int ct     = blockIdx.y;
  const int lane   = threadIdx.x;
  const int s      = chunk / nCper;
  const int cc     = chunk - s * nCper;
  const int m      = lane & 15;
  const int akb    = (lane >> 4) << 4;
  const int co     = ct * 16 + m;
  v16h f;
#pragma unroll
  for (int e = 0; e < 16; ++e) {
    int kl = akb + e;
    int ci = cc * 32 + kl;
    float v = 0.0f;
    if (ci < Cin) {
      if (mode == 2) v = W[(size_t)ci * Cout * 9 + co * 9 + (8 - s)];
      else           v = W[((size_t)co * Cin + ci) * ks2 + s];
    }
    f[e] = (_Float16)v;
  }
  ((v16h*)P)[((size_t)ct * nChunk + chunk) * 32 + lane] = f;
}

// ---------------------------------------------------------------------------
// Implicit-GEMM conv. One wave computes CT cout-tiles x 16 pixels (one row).
// Input Xh: f16 NHWC, channel stride = Cinp (zero padded).
//   MODE 0: zero-pad (stride/pad/dil), MODE 1: reflect-pad-1 k3 s1,
//   MODE 2: ConvTranspose2d(k3,s2,p1,op1) (weights pre-flipped).
//   act 0 none, 1 lrelu(0.2), 2 tanh.  Wo/H/W powers of two.
// Y: f32 NCHW (channel slab oct/ocf); Yh (optional): f16 NHWC stride oct.
// ---------------------------------------------------------------------------
template<int KS, int MODE, int CT>
__global__ __launch_bounds__(256) void conv_wmma_kernel(
    const _Float16* __restrict__ Xh, const v16h* __restrict__ Apack,
    const float* __restrict__ Bi, float* __restrict__ Y, _Float16* __restrict__ Yh,
    int Bn, int Cinp, int H, int Wd, int Cout, int Ho, int woShift,
    int stride, int pad, int dil, int act, int outCtot, int outCoff)
{
  constexpr int KS2 = KS * KS;
  const int Wo   = 1 << woShift;
  const int lane = threadIdx.x & 31;
  const int wid  = blockIdx.x * (blockDim.x >> 5) + (threadIdx.x >> 5);
  const int pixTiles = (Ho << woShift) >> 4;
  const int coGroups = Cout / (16 * CT);
  if (wid >= Bn * coGroups * pixTiles) return;   // wave-uniform exit

  int pt  = wid % pixTiles;
  int tmp = wid / pixTiles;
  int cg  = tmp % coGroups;
  int b   = tmp / coGroups;

  const int nCper  = Cinp >> 5;
  const int nChunk = KS2 * nCper;

  const int bkb = (lane >> 4) << 4;          // B K-base 0/16 per half-wave
  const int pix = (pt << 4) + (lane & 15);
  const int oh  = pix >> woShift;            // identical across the wave (Wo>=16)
  const int ow  = pix & (Wo - 1);
  const int ohs = oh * stride - pad;
  const int ows = ow * stride - pad;

  const _Float16* Xb = Xh + (size_t)b * H * Wd * Cinp;
  const v16h*     Ap = Apack + (size_t)(cg * CT) * nChunk * 32 + lane;

  v8f zero = {};
  v8f acc[CT];
#pragma unroll
  for (int t = 0; t < CT; ++t) acc[t] = zero;
  v16h bz;
#pragma unroll
  for (int e = 0; e < 16; ++e) bz[e] = (_Float16)0.0f;

#pragma unroll
  for (int s = 0; s < KS2; ++s) {
    const int kh = s / KS, kw = s - kh * KS;
    int ih, iw;
    bool rowOK = true, colOK = true;
    if constexpr (MODE == 0) {
      ih = ohs + kh * dil;
      iw = ows + kw * dil;
      rowOK = (unsigned)ih < (unsigned)H;
      colOK = (unsigned)iw < (unsigned)Wd;
    } else if constexpr (MODE == 1) {       // reflect pad 1: always in-bounds
      ih = oh + kh - 1; iw = ow + kw - 1;
      ih = ih < 0 ? -ih : (ih >= H  ? 2 * H  - 2 - ih : ih);
      iw = iw < 0 ? -iw : (iw >= Wd ? 2 * Wd - 2 - iw : iw);
    } else {                                 // convT: lhs_dilation 2, pad (1,2)
      int th = oh + kh - 1, tw = ow + kw - 1;
      ih = th >> 1; iw = tw >> 1;
      rowOK = (th >= 0) && !(th & 1) && (ih < H);
      colOK = (tw >= 0) && !(tw & 1) && (iw < Wd);
    }
    if constexpr (MODE != 1) {
      // row validity is wave-uniform -> scalar skip, EXEC untouched
      if (__builtin_amdgcn_readfirstlane((int)rowOK) == 0) continue;
    }
    const _Float16* bp =
        Xb + ((size_t)ih * Wd + (colOK ? iw : 0)) * Cinp + bkb;
    for (int cc = 0; cc < nCper; ++cc) {
      v16h bfv = *(const v16h*)(bp + (cc << 5));     // one aligned 32B load
      __builtin_prefetch(bp + (cc << 5) + 32, 0, 1); // next chunk's line
      if constexpr (MODE != 1) {
        bfv = colOK ? bfv : bz;
      }
      const v16h* ap = Ap + ((size_t)(s * nCper + cc) << 5);
#pragma unroll
      for (int t = 0; t < CT; ++t) {
        v16h af = ap[(size_t)t * nChunk * 32];       // aligned 32B fragment
        acc[t] = __builtin_amdgcn_wmma_f32_16x16x32_f16(
            false, af, false, bfv, (short)0, acc[t], false, false);
      }
    }
  }

  // D layout: VGPR r -> M = r + 8*(lane>=16), N = lane & 15
  const int n   = lane & 15;
  const int mb  = (lane >> 4) << 3;
  const int opix = (pt << 4) + n;
  const int ooh = opix >> woShift;
  const int oow = opix & (Wo - 1);
  const bool wantH = (Yh != nullptr);
#pragma unroll
  for (int t = 0; t < CT; ++t) {
    const int coB = (cg * CT + t) << 4;
#pragma unroll
    for (int r = 0; r < 8; ++r) {
      int co = coB + mb + r;
      float v = acc[t][r] + Bi[co];
      if (act == 1)      v = v > 0.0f ? v : 0.2f * v;
      else if (act == 2) v = tanhf(v);
      size_t oidx = ((((size_t)b * outCtot + outCoff + co) * Ho + ooh) << woShift) + oow;
      Y[oidx] = v;
      if (wantH)
        Yh[((size_t)b * (Ho << woShift) + opix) * outCtot + outCoff + co] = (_Float16)v;
    }
  }
}

// ---------------------------------------------------------------------------
// Block reduction (blockDim.x == 256, 8 wave32 waves)
// ---------------------------------------------------------------------------
__device__ __forceinline__ float blockReduceSum(float v) {
  __shared__ float sh[8];
  const int lane = threadIdx.x & 31;
  const int wrp  = threadIdx.x >> 5;
#pragma unroll
  for (int o = 16; o > 0; o >>= 1) v += __shfl_down(v, o, 32);
  if (lane == 0) sh[wrp] = v;
  __syncthreads();
  float r = (threadIdx.x < 8) ? sh[threadIdx.x] : 0.0f;
  if (wrp == 0) {
#pragma unroll
    for (int o = 4; o > 0; o >>= 1) r += __shfl_down(r, o, 32);
    if (lane == 0) sh[0] = r;
  }
  __syncthreads();
  r = sh[0];
  __syncthreads();
  return r;
}

// Instance norm; optional affine; optional lrelu; optional f16 NHWC mirror.
__global__ __launch_bounds__(256) void inorm_kernel(
    const float* __restrict__ X, float* __restrict__ Y, _Float16* __restrict__ Yh,
    const float* __restrict__ g, const float* __restrict__ bt,
    int C, int HW, int act)
{
  const int bc = blockIdx.x;
  const int c  = bc % C;
  const int b  = bc / C;
  const float* xp = X + (size_t)bc * HW;
  float*       yp = Y + (size_t)bc * HW;
  float s = 0.0f, ss = 0.0f;
  for (int i = threadIdx.x; i < HW; i += 256) { float v = xp[i]; s += v; ss += v * v; }
  s  = blockReduceSum(s);
  ss = blockReduceSum(ss);
  const float mu  = s / (float)HW;
  const float var = fmaxf(ss / (float)HW - mu * mu, 0.0f);
  const float inv = rsqrtf(var + 1e-5f);
  const float gg  = g  ? g[c]  : 1.0f;
  const float bb  = bt ? bt[c] : 0.0f;
  const bool wantH = (Yh != nullptr);
  _Float16* yh = Yh + ((size_t)b * HW) * C + c;   // NHWC, stride C
  for (int i = threadIdx.x; i < HW; i += 256) {
    float v = (xp[i] - mu) * inv * gg + bb;
    if (act) v = v > 0.0f ? v : 0.2f * v;
    yp[i] = v;
    if (wantH) yh[(size_t)i * C] = (_Float16)v;
  }
}

// x [4,3,256,256] f32 -> NHWC f16 padded to 32 channels: [4, 65536, 32]
__global__ void x_to_nhwc_kernel(const float* __restrict__ x, _Float16* __restrict__ y, int n) {
  int i = blockIdx.x * blockDim.x + threadIdx.x;
  if (i >= n) return;
  int cp = i & 31;
  int hw = (i >> 5) & 65535;
  int b  = i >> 21;
  y[i] = (cp < 3) ? (_Float16)x[(((size_t)b * 3 + cp) << 16) + hw] : (_Float16)0.0f;
}

// residual: tensors [4,256,1024]; yh NHWC [4,1024,256]
__global__ void add_res_kernel(const float* __restrict__ a, const float* __restrict__ b,
                               float* __restrict__ y, _Float16* __restrict__ yh, int n) {
  int i = blockIdx.x * blockDim.x + threadIdx.x;
  if (i >= n) return;
  float v = a[i] + b[i];
  y[i] = v;
  int hw = i & 1023;
  int c  = (i >> 10) & 255;
  int bn = i >> 18;
  yh[((((size_t)bn << 10) + hw) << 8) + c] = (_Float16)v;
}

// y[b,c,hw] = cat[b, slab*64+c, hw] + b2[b,c,hw]; y [4,64,1024], yh NHWC [4,1024,64]
__global__ void add_cat_kernel(const float* __restrict__ cat, int slab,
                               const float* __restrict__ b2, float* __restrict__ y,
                               _Float16* __restrict__ yh, int n) {
  int i = blockIdx.x * blockDim.x + threadIdx.x;
  if (i >= n) return;
  int hw = i & 1023;
  int c  = (i >> 10) & 63;
  int bn = i >> 16;
  float v = cat[(((size_t)bn * 256 + slab * 64 + c) << 10) + hw] + b2[i];
  y[i] = v;
  yh[((((size_t)bn << 10) + hw) << 6) + c] = (_Float16)v;
}

// cv[b,n,c] = masked mean of codes[b,c,128,128] over segmap[b,n,256,256] (2x nearest)
__global__ __launch_bounds__(256) void segmean_kernel(
    const float* __restrict__ codes, const float* __restrict__ seg, float* __restrict__ cv)
{
  const int idx = blockIdx.x;            // b*10*512 + n*512 + c
  const int c   = idx & 511;
  const int t   = idx >> 9;
  const int ncl = t % 10;
  const int b   = t / 10;
  const float* cp = codes + ((size_t)((b << 9) + c)) * 16384;
  const float* sp = seg   + ((size_t)(b * 10 + ncl)) * 65536;
  float s = 0.0f, a = 0.0f;
  for (int i = threadIdx.x; i < 16384; i += 256) {
    int oh = i >> 7, ow = i & 127;
    float m = (sp[(oh << 9) + (ow << 1)] != 0.0f) ? 1.0f : 0.0f;   // (2*oh)*256 + 2*ow
    s += cp[i] * m;
    a += m;
  }
  s = blockReduceSum(s);
  a = blockReduceSum(a);
  if (threadIdx.x == 0)
    cv[idx] = (a > 0.0f) ? s / fmaxf(a, 1.0f) : 0.0f;
}

// ---------------------------------------------------------------------------
// Host-side orchestration
// ---------------------------------------------------------------------------
extern "C" void kernel_launch(void* const* d_in, const int* in_sizes, int n_in,
                              void* d_out, int out_size, void* d_ws, size_t ws_size,
                              hipStream_t stream) {
  (void)in_sizes; (void)n_in; (void)out_size; (void)ws_size;
  const float* x    = (const float*)d_in[0];
  const float* seg  = (const float*)d_in[1];
  const float* w0   = (const float*)d_in[2];  const float* b0   = (const float*)d_in[3];
  const float* wd1  = (const float*)d_in[4];  const float* bd1  = (const float*)d_in[5];
  const float* wd2  = (const float*)d_in[6];  const float* bd2  = (const float*)d_in[7];
  const float* wd3  = (const float*)d_in[8];  const float* bd3  = (const float*)d_in[9];
  const float* dw3  = (const float*)d_in[10]; const float* db3  = (const float*)d_in[11];
  const float* dw1  = (const float*)d_in[12]; const float* db1  = (const float*)d_in[13];
  const float* dw2  = (const float*)d_in[14]; const float* db2  = (const float*)d_in[15];
  const float* dw4  = (const float*)d_in[16]; const float* db4  = (const float*)d_in[17];
  const float* dw8  = (const float*)d_in[18]; const float* db8  = (const float*)d_in[19];
  const float* dw12 = (const float*)d_in[20]; const float* db12 = (const float*)d_in[21];
  const float* dw24 = (const float*)d_in[22]; const float* db24 = (const float*)d_in[23];
  const float* dw48 = (const float*)d_in[24]; const float* db48 = (const float*)d_in[25];
  const float* dg   = (const float*)d_in[26]; const float* dbt  = (const float*)d_in[27];
  const float* dwc  = (const float*)d_in[28]; const float* dbc  = (const float*)d_in[29];
  const float* dgc  = (const float*)d_in[30]; const float* dbtc = (const float*)d_in[31];
  const float* we   = (const float*)d_in[32]; const float* be   = (const float*)d_in[33];
  const float* wt1  = (const float*)d_in[34]; const float* bt1  = (const float*)d_in[35];
  const float* wt2  = (const float*)d_in[36]; const float* bt2  = (const float*)d_in[37];
  const float* wf   = (const float*)d_in[38]; const float* bf   = (const float*)d_in[39];

  float* ws = (float*)d_ws;
  // f32 regions (floats):
  //   [0, 8.39M)         t0 [4,32,256,256], later c1 [4,512,64,64]
  //   [8.39M, 41.94M)    encoder chain, later codes [4,512,128,128]
  //   [41.94M, 109.05M)  c2 [4,1024,128,128]
  //   [109.05M, 115.45M) packed f16 weights
  //   [115.45M, ~166.6M) f16 NHWC activation mirrors (~666 MB total ws)
  float* t0   = ws;
  float* c1   = t0;
  float* t1   = ws + 8388608;
  float* t2   = t1 + 4194304;
  float* mid  = t2 + 2097152;
  float* x1b  = mid + 1048576;
  float* cat  = x1b + 262144;
  float* x22  = cat + 1048576;
  float* x24  = x22 + 262144;
  float* x28  = x24 + 262144;
  float* tmpA = x28 + 262144;
  float* x2b  = tmpA + 262144;
  float* dmb  = x2b + 1048576;
  float* t3   = dmb + 1048576;
  float* codes = ws + 8388608;
  float* c2   = ws + 8388608 + 33554432;
  _Float16* pw = (_Float16*)(ws + 109051904);
  _Float16* mh = (_Float16*)(ws + 115451904);

  float* enc = (float*)d_out;             // [4,512,16,16] = 524288
  float* cv  = enc + 4 * 512 * 16 * 16;   // [4,10,512]    = 20480

  // f16 NHWC activation mirrors
  auto halloc = [&](size_t n) { _Float16* p = mh; mh += n; return p; };
  _Float16* xh   = halloc(8388608);   // [4,65536,32] (3 ch padded to 32)
  _Float16* t0h  = halloc(8388608);   // [4,65536,32]
  _Float16* t1h  = halloc(4194304);   // [4,16384,64]
  _Float16* t2h  = halloc(2097152);   // [4,4096,128]
  _Float16* midh = halloc(1048576);   // [4,1024,256]
  _Float16* x1bh = halloc(262144);    // [4,1024,64]
  _Float16* cath = halloc(1048576);   // [4,1024,256]
  _Float16* tAh  = halloc(262144);    // [4,1024,64]
  _Float16* dmbh = halloc(1048576);   // [4,1024,256]
  _Float16* c1h  = halloc(8388608);   // [4,4096,512]
  _Float16* c2h  = halloc(67108864);  // [4,16384,1024]

  // --- pack conv weights into WMMA A-fragment order ---
  auto pack = [&](const float* W, int Cout, int Cin, int ks, int mode) -> _Float16* {
    int Cinp = (Cin + 31) & ~31;
    int nChunk = ks * ks * (Cinp / 32);
    _Float16* p = pw;
    pw += (size_t)Cout * nChunk * 32;
    pack_w_kernel<<<dim3(nChunk, Cout / 16), dim3(32), 0, stream>>>(W, p, Cout, Cin, ks, mode);
    return p;
  };
  _Float16* p_w0  = pack(w0,  32,   3,    3, 1);
  _Float16* p_wd1 = pack(wd1, 64,   32,   3, 0);
  _Float16* p_wd2 = pack(wd2, 128,  64,   3, 0);
  _Float16* p_wd3 = pack(wd3, 256,  128,  3, 0);
  _Float16* p_dw3 = pack(dw3, 64,   256,  3, 0);
  _Float16* p_dw1 = pack(dw1, 64,   64,   3, 0);
  _Float16* p_dw2 = pack(dw2, 64,   64,   3, 0);
  _Float16* p_dw4 = pack(dw4, 64,   64,   3, 0);
  _Float16* p_dw8 = pack(dw8, 64,   64,   3, 0);
  _Float16* p_d12 = pack(dw12,64,   64,   3, 0);
  _Float16* p_d24 = pack(dw24,64,   64,   3, 0);
  _Float16* p_d48 = pack(dw48,64,   64,   3, 0);
  _Float16* p_dwc = pack(dwc, 256,  256,  1, 0);
  _Float16* p_we  = pack(we,  512,  256,  3, 0);
  _Float16* p_wt1 = pack(wt1, 512,  256,  3, 2);
  _Float16* p_wt2 = pack(wt2, 1024, 512,  3, 2);
  _Float16* p_wf  = pack(wf,  512,  1024, 3, 1);

  auto conv = [&](const _Float16* Xi, const _Float16* P, const float* Bi,
                  float* Y, _Float16* Yh,
                  int Cin, int H, int Wd, int Cout, int Ho, int Wo,
                  int ks, int st, int pd, int dl, int mode, int act,
                  int oct, int ocf) {
    int woShift = __builtin_ctz(Wo);
    int Cinp = (Cin + 31) & ~31;
    const int CT = (Cout % 64 == 0) ? 4 : 2;
    long waves = 4L * (Cout / (16 * CT)) * ((long)(Ho * Wo) / 16);
    unsigned blocks = (unsigned)((waves + 7) / 8);
    dim3 g(blocks), bk(256);
#define CARGS Xi, (const v16h*)P, Bi, Y, Yh, 4, Cinp, H, Wd, Cout, Ho, woShift, st, pd, dl, act, oct, ocf
    if (ks == 1)                   conv_wmma_kernel<1, 0, 4><<<g, bk, 0, stream>>>(CARGS);
    else if (mode == 0)            conv_wmma_kernel<3, 0, 4><<<g, bk, 0, stream>>>(CARGS);
    else if (mode == 1 && CT == 2) conv_wmma_kernel<3, 1, 2><<<g, bk, 0, stream>>>(CARGS);
    else if (mode == 1)            conv_wmma_kernel<3, 1, 4><<<g, bk, 0, stream>>>(CARGS);
    else                           conv_wmma_kernel<3, 2, 4><<<g, bk, 0, stream>>>(CARGS);
#undef CARGS
  };
  auto inorm = [&](const float* X, float* Y, _Float16* Yh, const float* g, const float* bt,
                   int C, int HW, int act) {
    inorm_kernel<<<dim3(4 * C), dim3(256), 0, stream>>>(X, Y, Yh, g, bt, C, HW, act);
  };

  // x -> padded NHWC f16 mirror
  x_to_nhwc_kernel<<<dim3(8388608 / 256), dim3(256), 0, stream>>>(x, xh, 8388608);

  // ---- encoder trunk ----
  conv(xh, p_w0, b0, t0, nullptr, 3, 256, 256, 32, 256, 256, 3, 1, 1, 1, 1, 0, 32, 0);
  inorm(t0, t0, t0h, nullptr, nullptr, 32, 65536, 1);
  conv(t0h, p_wd1, bd1, t1, nullptr, 32, 256, 256, 64, 128, 128, 3, 2, 1, 1, 0, 0, 64, 0);
  inorm(t1, t1, t1h, nullptr, nullptr, 64, 16384, 1);
  conv(t1h, p_wd2, bd2, t2, nullptr, 64, 128, 128, 128, 64, 64, 3, 2, 1, 1, 0, 0, 128, 0);
  inorm(t2, t2, t2h, nullptr, nullptr, 128, 4096, 1);
  conv(t2h, p_wd3, bd3, mid, nullptr, 128, 64, 64, 256, 32, 32, 3, 2, 1, 1, 0, 0, 256, 0);
  inorm(mid, mid, midh, nullptr, nullptr, 256, 1024, 1);

  // ---- DMFB ----
  conv(midh, p_dw3, db3, x1b, nullptr, 256, 32, 32, 64, 32, 32, 3, 1, 1, 1, 0, 0, 64, 0);
  inorm(x1b, x1b, x1bh, dg, dbt, 64, 1024, 1);
  conv(x1bh, p_dw1, db1, cat, cath, 64, 32, 32, 64, 32, 32, 3, 1, 1, 1, 0, 1, 256, 0);     // x2_1
  conv(x1bh, p_dw2, db2, x22, nullptr, 64, 32, 32, 64, 32, 32, 3, 1, 2, 2, 0, 1, 64, 0);
  conv(x1bh, p_dw4, db4, x24, nullptr, 64, 32, 32, 64, 32, 32, 3, 1, 4, 4, 0, 1, 64, 0);
  conv(x1bh, p_dw8, db8, x28, nullptr, 64, 32, 32, 64, 32, 32, 3, 1, 8, 8, 0, 1, 64, 0);
  int nadd = 4 * 64 * 1024;
  add_cat_kernel<<<dim3(nadd / 256), dim3(256), 0, stream>>>(cat, 0, x22, tmpA, tAh, nadd);
  conv(tAh, p_d12, db12, cat, cath, 64, 32, 32, 64, 32, 32, 3, 1, 1, 1, 0, 0, 256, 64);    // k2
  add_cat_kernel<<<dim3(nadd / 256), dim3(256), 0, stream>>>(cat, 1, x24, tmpA, tAh, nadd);
  conv(tAh, p_d24, db24, cat, cath, 64, 32, 32, 64, 32, 32, 3, 1, 1, 1, 0, 0, 256, 128);   // k3
  add_cat_kernel<<<dim3(nadd / 256), dim3(256), 0, stream>>>(cat, 2, x28, tmpA, tAh, nadd);
  conv(tAh, p_d48, db48, cat, cath, 64, 32, 32, 64, 32, 32, 3, 1, 1, 1, 0, 0, 256, 192);   // k4
  conv(cath, p_dwc, dbc, x2b, nullptr, 256, 32, 32, 256, 32, 32, 1, 1, 0, 1, 0, 0, 256, 0);
  inorm(x2b, x2b, nullptr, dgc, dbtc, 256, 1024, 0);
  int nres = 4 * 256 * 1024;
  add_res_kernel<<<dim3(nres / 256), dim3(256), 0, stream>>>(x2b, mid, dmb, dmbh, nres);

  // ---- enc head (output 0) ----
  conv(dmbh, p_we, be, t3, nullptr, 256, 32, 32, 512, 16, 16, 3, 2, 1, 1, 0, 0, 512, 0);
  inorm(t3, enc, nullptr, nullptr, nullptr, 512, 256, 1);

  // ---- decoder / codes ----
  conv(midh, p_wt1, bt1, c1, nullptr, 256, 32, 32, 512, 64, 64, 3, 1, 1, 1, 2, 0, 512, 0);
  inorm(c1, c1, c1h, nullptr, nullptr, 512, 4096, 1);
  conv(c1h, p_wt2, bt2, c2, nullptr, 512, 64, 64, 1024, 128, 128, 3, 1, 1, 1, 2, 0, 1024, 0);
  inorm(c2, c2, c2h, nullptr, nullptr, 1024, 16384, 1);
  conv(c2h, p_wf, bf, codes, nullptr, 1024, 128, 128, 512, 128, 128, 3, 1, 1, 1, 1, 2, 512, 0);

  // ---- segment mean (output 1) ----
  segmean_kernel<<<dim3(4 * 10 * 512), dim3(256), 0, stream>>>(codes, seg, cv);
}
